// GraphTestIn_14877766713833
// MI455X (gfx1250) — compile-verified, matrix operand order checked by reading, not run on hardware
//
#include <hip/hip_runtime.h>
#include <hip/hip_bf16.h>

typedef __attribute__((ext_vector_type(16))) _Float16     v16h;
typedef __attribute__((ext_vector_type(8)))  float        v8f;
typedef __attribute__((ext_vector_type(8)))  unsigned int v8u;

#define NN   512
#define EE   (NN * NN)
#define NVF  6
#define WPB  4          // waves per block (wave32)

static __device__ __forceinline__ unsigned pack2h(float a, float b)
{
    union { _Float16 h[2]; unsigned u; } p;
    p.h[0] = (_Float16)a;
    p.h[1] = (_Float16)b;
    return p.u;
}

// ---------------------------------------------------------------------------
// Degree histogram: cnt[dst] += 1
// ---------------------------------------------------------------------------
__global__ void deg_count(const long long* __restrict__ edge_index,
                          float* __restrict__ cnt)
{
    int e = blockIdx.x * blockDim.x + threadIdx.x;
    if (e < EE) atomicAdd(&cnt[(int)edge_index[EE + e]], 1.0f);
}

// ---------------------------------------------------------------------------
// Fused NNConv edge kernel (WMMA).
// One wave32 per 16-edge tile:
//   t = relu(edge_attr @ W + b)            via v_wmma_f32_16x16x32_f16
//   msg[e,o] = sum_i x[src[e]][i] * t[i,o] fused per input channel i
//   atomicAdd(sum[dst[e]][o], msg[e,o])
// A: lanes 0-15 = edge rows, K=0..5 in elements 0..5 (rest zero).
// B: lanes 0-15 hold K=0..5 in elements 0..5 (rest zero), N = lane; lanes 16-31 read
//    a zero slot (their K range 8..15, 24..31 is entirely padding).
// D: VGPR r -> (M=r, N=lane) lanes 0-15; (M=8+r, N=lane-16) lanes 16-31.
// B is pre-packed in LDS as uint4 per (i, chunk, column): 3 f16-pairs + pad,
// so the operand build is a single aligned ds_load_b128.
// ---------------------------------------------------------------------------
template<int IN, int OUT, int OUTP>
__global__ __launch_bounds__(WPB * 32)
void nnconv_edge_wmma(const float* __restrict__ hprev,         // [NN, IN]
                      const float* __restrict__ edge_attr,     // [EE, 6]
                      const long long* __restrict__ edge_index,// [2, EE]
                      const float* __restrict__ W,             // [6, IN*OUT]
                      const float* __restrict__ bnn,           // [IN*OUT]
                      float* __restrict__ sum_out)             // [NN, OUT]
{
    constexpr int NCH = OUTP / 16;
    constexpr int NWP = IN * NCH * 16;      // packed B entries
    __shared__ uint4 Wp[NWP + 1];           // +1 = zero slot for hi lanes
    __shared__ float bldsp[IN * OUTP];      // bias, zero-padded to OUTP
    __shared__ __align__(16) float xsT[WPB][IN * 16];  // x[src], [i][M]
    __shared__ int dsts[WPB][16];

    const int tid  = threadIdx.x;
    const int lane = tid & 31;
    const int wave = tid >> 5;
    const int row  = lane & 15;
    const bool lo  = lane < 16;

    // ---- Stage packed-f16 B tiles, padded bias, zero slot ----
    for (int idx = tid; idx < NWP; idx += blockDim.x) {
        const int n  = idx & 15;
        const int ih = idx >> 4;
        const int i  = ih / NCH;
        const int h  = ih - i * NCH;
        const int colo = h * 16 + n;
        uint4 w = make_uint4(0u, 0u, 0u, 0u);
        if (colo < OUT) {
            const float* wc = W + i * OUT + colo;   // stride IN*OUT per k
            w.x = pack2h(wc[0 * (IN * OUT)], wc[1 * (IN * OUT)]);
            w.y = pack2h(wc[2 * (IN * OUT)], wc[3 * (IN * OUT)]);
            w.z = pack2h(wc[4 * (IN * OUT)], wc[5 * (IN * OUT)]);
        }
        Wp[idx] = w;
    }
    if (tid == 0) Wp[NWP] = make_uint4(0u, 0u, 0u, 0u);
    for (int idx = tid; idx < IN * OUTP; idx += blockDim.x) {
        const int i = idx / OUTP;
        const int o = idx - i * OUTP;
        bldsp[idx] = (o < OUT) ? bnn[i * OUT + o] : 0.0f;
    }
    __syncthreads();

    const long long* srcI = edge_index;
    const long long* dstI = edge_index + EE;

    const int tile = blockIdx.x * WPB + wave;
    const int e0   = tile * 16;

    // ---- A operand: edge_attr tile, K padded 6 -> 32 with zeros ----
    v16h a;
    #pragma unroll
    for (int k = 0; k < 16; ++k) a[k] = (_Float16)0.0f;
    if (lo) {
        const float* er = edge_attr + (size_t)(e0 + row) * NVF;
        #pragma unroll
        for (int k = 0; k < NVF; ++k) a[k] = (_Float16)er[k];
    }

    // ---- Stage x[src] rows (transposed: [i][M]) and dst ids ----
    for (int idx = lane; idx < 16 * IN; idx += 32) {
        const int r = idx / IN;
        const int i = idx - r * IN;
        const int s = (int)srcI[e0 + r];
        xsT[wave][i * 16 + r] = hprev[s * IN + i];
    }
    if (lo) dsts[wave][row] = (int)dstI[e0 + row];
    __syncthreads();

    float macc[NCH][8];
    #pragma unroll
    for (int h = 0; h < NCH; ++h)
        #pragma unroll
        for (int r = 0; r < 8; ++r) macc[h][r] = 0.0f;

    const int  mbase = lo ? 0 : 8;
    const int  colb  = lane & 15;
    const v8f  zc    = {};   // inline-0 SRC2

    for (int i = 0; i < IN; ++i) {
        // x[src] values for this input channel: two aligned b128 loads
        const float4* xp = (const float4*)&xsT[wave][i * 16 + mbase];
        const float4 xa = xp[0];
        const float4 xb = xp[1];
        const float xv[8] = { xa.x, xa.y, xa.z, xa.w,
                              xb.x, xb.y, xb.z, xb.w };

        v8f   cc[NCH];
        float bb[NCH];
        #pragma unroll
        for (int h = 0; h < NCH; ++h) {
            const int widx = lo ? ((i * NCH + h) * 16 + lane) : NWP;
            const uint4 wv = Wp[widx];           // ds_load_b128
            v8u u;
            u[0] = wv.x; u[1] = wv.y; u[2] = wv.z; u[3] = 0u;
            u[4] = 0u;   u[5] = 0u;   u[6] = 0u;  u[7] = 0u;
            const v16h bv = __builtin_bit_cast(v16h, u);
            bb[h] = bldsp[i * OUTP + h * 16 + colb];
            cc[h] = __builtin_amdgcn_wmma_f32_16x16x32_f16(
                        false, a, false, bv, (short)0, zc, false, false);
        }
        #pragma unroll
        for (int h = 0; h < NCH; ++h) {
            #pragma unroll
            for (int r = 0; r < 8; ++r) {
                float t = cc[h][r] + bb[h];       // bias inside the relu
                t = t > 0.0f ? t : 0.0f;
                macc[h][r] = fmaf(xv[r], t, macc[h][r]);
            }
        }
    }

    // ---- Scatter msg into per-dst sums ----
    #pragma unroll
    for (int h = 0; h < NCH; ++h) {
        const int colo = h * 16 + colb;
        if (colo < OUT) {
            #pragma unroll
            for (int r = 0; r < 8; ++r) {
                const int d = dsts[wave][mbase + r];
                atomicAdd(&sum_out[d * OUT + colo], macc[h][r]);
            }
        }
    }
}

// ---------------------------------------------------------------------------
// Combine: h_out = relu( sum/max(cnt,1) + hprev @ root + bias )
// ---------------------------------------------------------------------------
template<int IN, int OUT>
__global__ void nnconv_combine(const float* __restrict__ summ,
                               const float* __restrict__ cnt,
                               const float* __restrict__ hprev,
                               const float* __restrict__ root,
                               const float* __restrict__ bias,
                               float* __restrict__ hout)
{
    int idx = blockIdx.x * blockDim.x + threadIdx.x;
    if (idx >= NN * OUT) return;
    const int n = idx / OUT;
    const int o = idx - n * OUT;
    float c = cnt[n];
    c = c > 1.0f ? c : 1.0f;
    float v = summ[idx] / c + bias[o];
    #pragma unroll
    for (int i = 0; i < IN; ++i)
        v = fmaf(hprev[n * IN + i], root[i * OUT + o], v);
    hout[idx] = v > 0.0f ? v : 0.0f;
}

// ---------------------------------------------------------------------------
// CBT: out[a,b] = sum_f |h3[b,f] - h3[a,f]|
// ---------------------------------------------------------------------------
__global__ void cbt_kernel(const float* __restrict__ h3, float* __restrict__ out)
{
    const int a = blockIdx.x;
    float ha[5];
    #pragma unroll
    for (int f = 0; f < 5; ++f) ha[f] = h3[a * 5 + f];
    for (int b = threadIdx.x; b < NN; b += blockDim.x) {
        float s = 0.0f;
        #pragma unroll
        for (int f = 0; f < 5; ++f) s += fabsf(h3[b * 5 + f] - ha[f]);
        out[a * NN + b] = s;
    }
}

// ---------------------------------------------------------------------------
extern "C" void kernel_launch(void* const* d_in, const int* in_sizes, int n_in,
                              void* d_out, int out_size, void* d_ws, size_t ws_size,
                              hipStream_t stream)
{
    const float*     x     = (const float*)d_in[0];
    const float*     ea    = (const float*)d_in[1];
    const long long* ei    = (const long long*)d_in[2];
    const float*     W1    = (const float*)d_in[3];
    const float*     b1    = (const float*)d_in[4];
    const float*     root1 = (const float*)d_in[5];
    const float*     bias1 = (const float*)d_in[6];
    const float*     W2    = (const float*)d_in[7];
    const float*     b2    = (const float*)d_in[8];
    const float*     root2 = (const float*)d_in[9];
    const float*     bias2 = (const float*)d_in[10];
    const float*     W3    = (const float*)d_in[11];
    const float*     b3    = (const float*)d_in[12];
    const float*     root3 = (const float*)d_in[13];
    const float*     bias3 = (const float*)d_in[14];
    float* out = (float*)d_out;

    float* ws   = (float*)d_ws;
    float* cnt  = ws;                 // NN
    float* sum1 = cnt  + NN;          // NN*36
    float* h1   = sum1 + NN * 36;     // NN*36
    float* sum2 = h1   + NN * 36;     // NN*24
    float* h2   = sum2 + NN * 24;     // NN*24
    float* sum3 = h2   + NN * 24;     // NN*5
    float* h3   = sum3 + NN * 5;      // NN*5
    const size_t used = (size_t)NN * (1 + 36 + 36 + 24 + 24 + 5 + 5);

    hipMemsetAsync(d_ws, 0, used * sizeof(float), stream);

    deg_count<<<EE / 256, 256, 0, stream>>>(ei, cnt);

    const int eblocks = (EE / 16) / WPB;   // 4096 blocks, 4 wave32 each

    // Layer 1: 1 -> 36  (OUTP = 48 -> 3 WMMA chunks per input channel)
    nnconv_edge_wmma<1, 36, 48><<<eblocks, WPB * 32, 0, stream>>>(
        x, ea, ei, W1, b1, sum1);
    nnconv_combine<1, 36><<<(NN * 36 + 255) / 256, 256, 0, stream>>>(
        sum1, cnt, x, root1, bias1, h1);

    // Layer 2: 36 -> 24 (OUTP = 32 -> 2 chunks; 72 WMMAs per 16-edge tile)
    nnconv_edge_wmma<36, 24, 32><<<eblocks, WPB * 32, 0, stream>>>(
        h1, ea, ei, W2, b2, sum2);
    nnconv_combine<36, 24><<<(NN * 24 + 255) / 256, 256, 0, stream>>>(
        sum2, cnt, h1, root2, bias2, h2);

    // Layer 3: 24 -> 5  (OUTP = 16 -> 1 chunk)
    nnconv_edge_wmma<24, 5, 16><<<eblocks, WPB * 32, 0, stream>>>(
        h2, ea, ei, W3, b3, sum3);
    nnconv_combine<24, 5><<<(NN * 5 + 255) / 256, 256, 0, stream>>>(
        sum3, cnt, h2, root3, bias3, h3);

    cbt_kernel<<<NN, 256, 0, stream>>>(h3, out);
}
